// ScaledDotProductAttention__22986664969071
// MI455X (gfx1250) — matrix operand order repaired and checked by reading, not compiled, not run on hardware
//
#include <hip/hip_runtime.h>
#include <stdint.h>

// ---------------------------------------------------------------------------
// Types (names chosen to avoid HIP's predefined ushortN vector types)
// ---------------------------------------------------------------------------
typedef __attribute__((ext_vector_type(16))) __bf16          bf16x16;
typedef __attribute__((ext_vector_type(16))) unsigned short  u16x16;
typedef __attribute__((ext_vector_type(8)))  unsigned short  u16x8;
typedef __attribute__((ext_vector_type(4)))  unsigned short  u16x4;
typedef __attribute__((ext_vector_type(8)))  float           v8f;
typedef __attribute__((ext_vector_type(4)))  float           v4f;

__device__ __forceinline__ unsigned short f32_to_bf16(float f) {
    uint32_t u = __builtin_bit_cast(uint32_t, f);
    u += 0x7FFFu + ((u >> 16) & 1u);   // round-to-nearest-even
    return (unsigned short)(u >> 16);
}

__device__ __forceinline__ bf16x16 combine_bf16(u16x8 lo, u16x8 hi) {
    u16x16 t;
#pragma unroll
    for (int j = 0; j < 8; ++j) { t[j] = lo[j]; t[8 + j] = hi[j]; }
    return __builtin_bit_cast(bf16x16, t);
}

__device__ __forceinline__ v8f wmma_bf16(bf16x16 a, bf16x16 b, v8f c) {
    return __builtin_amdgcn_wmma_f32_16x16x32_bf16(
        /*neg_a=*/false, a, /*neg_b=*/false, b,
        /*c_mod=*/(short)0, c, /*reuse_a=*/false, /*reuse_b=*/false);
}

// Async DMA: 16 bytes per lane, global -> LDS, tracked by ASYNCcnt.
__device__ __forceinline__ void async_load_b128(uint32_t lds_off, const void* gaddr) {
    asm volatile("global_load_async_to_lds_b128 %0, %1, off"
                 :: "v"(lds_off), "v"(gaddr)
                 : "memory");
}
__device__ __forceinline__ void wait_asynccnt0() {
    asm volatile("s_wait_asynccnt 0x0" ::: "memory");
}
__device__ __forceinline__ uint32_t lds_offset_of(const void* p) {
    // LDS flat aperture: low 32 bits of the flat address are the LDS byte offset
    return (uint32_t)(uintptr_t)p;
}

// ---------------------------------------------------------------------------
// Weight convert + transpose: Wt[n][k] = bf16(W[k][n]),  1024 x 1024
// (done once; lets GEMM B-tiles be straight async DMA copies)
// ---------------------------------------------------------------------------
__global__ __launch_bounds__(256) void cvt_w_transpose_kernel(
    const float* __restrict__ in, unsigned short* __restrict__ out) {
    int n = blockIdx.x * 32 + (threadIdx.x & 31);
    int k = blockIdx.y * 8 + (threadIdx.x >> 5);
    out[(size_t)n * 1024 + k] = f32_to_bf16(in[(size_t)k * 1024 + n]);
}

// ---------------------------------------------------------------------------
// Tiled bf16 WMMA GEMM:  C[M,N] = A[M,K] * Wt[N,K]^T + bias[N]
//   Wt is the PRE-TRANSPOSED bf16 weight (N x K row-major)
//   A_F32:    A is fp32 (converted to bf16 while staging to LDS),
//             else bf16 and staged via global_load_async_to_lds_b128
//   OUT_BF16: C stored as bf16, else fp32
// Block tile: 128(M) x 128(N), BK = 32, 256 threads = 8 waves,
// each wave computes a 32x64 sub-tile = 2x4 WMMA accumulators.
// ---------------------------------------------------------------------------
template <bool A_F32, bool OUT_BF16>
__global__ __launch_bounds__(256) void gemm_bf16_kernel(
    const void* __restrict__ Av, const unsigned short* __restrict__ Wt,
    const float* __restrict__ bias, void* __restrict__ Cv,
    int M, int N, int K) {
    __shared__ unsigned short Atile[128 * 32];  // [m][k]
    __shared__ unsigned short Btile[128 * 32];  // [n][k]

    const int tid  = threadIdx.x;
    const int wid  = tid >> 5;
    const int lane = tid & 31;
    const int half = lane >> 4;   // 0: lanes 0-15, 1: lanes 16-31
    const int l16  = lane & 15;

    const int bm = blockIdx.x * 128;
    const int bn = blockIdx.y * 128;
    const int wm = (wid & 3) * 32;   // wave M offset inside block tile
    const int wn = (wid >> 2) * 64;  // wave N offset inside block tile

    const uint32_t a_base = lds_offset_of(&Atile[0]);
    const uint32_t b_base = lds_offset_of(&Btile[0]);

    v8f acc[2][4];
#pragma unroll
    for (int i = 0; i < 2; ++i)
#pragma unroll
        for (int j = 0; j < 4; ++j) acc[i][j] = (v8f){0.f,0.f,0.f,0.f,0.f,0.f,0.f,0.f};

    const int nk = K / 32;
    for (int kt = 0; kt < nk; ++kt) {
        // ---- stage A tile (128 x 32 bf16) into LDS ----------------------
        if (A_F32) {
            const float* A = (const float*)Av;
#pragma unroll
            for (int p = 0; p < 4; ++p) {
                int row  = p * 32 + (tid >> 3);
                int col4 = (tid & 7) * 4;
                v4f f = *(const v4f*)(A + (size_t)(bm + row) * K + kt * 32 + col4);
                u16x4 h;
#pragma unroll
                for (int j = 0; j < 4; ++j) h[j] = f32_to_bf16(f[j]);
                *(u16x4*)&Atile[row * 32 + col4] = h;
            }
            if (kt + 1 < nk)
                __builtin_prefetch(A + (size_t)(bm + (tid >> 3)) * K + (kt + 1) * 32, 0, 1);
        } else {
            // bf16 row-major A: pure DMA copy, 2 x b128 per thread
            const unsigned short* A = (const unsigned short*)Av;
#pragma unroll
            for (int p = 0; p < 2; ++p) {
                int row = tid >> 1;
                int c   = (tid & 1) * 2 + p;   // 16-byte chunk 0..3 within row
                async_load_b128(a_base + (uint32_t)(row * 32 + c * 8) * 2,
                                A + (size_t)(bm + row) * K + kt * 32 + c * 8);
            }
        }
        // ---- stage B tile (128 n x 32 k bf16) via async DMA -------------
#pragma unroll
        for (int p = 0; p < 2; ++p) {
            int n = tid >> 1;
            int c = (tid & 1) * 2 + p;
            async_load_b128(b_base + (uint32_t)(n * 32 + c * 8) * 2,
                            Wt + (size_t)(bn + n) * K + kt * 32 + c * 8);
        }
        wait_asynccnt0();
        __syncthreads();

        // ---- build fragments and issue WMMAs ----------------------------
        // A fragment (16-bit A layout): lanes 0-15 -> K {0..7,16..23},
        // lanes 16-31 -> K {8..15,24..31}, row = l16.
        bf16x16 afr[2];
#pragma unroll
        for (int i = 0; i < 2; ++i) {
            int m    = wm + i * 16 + l16;
            int koff = half * 8;
            u16x8 lo = *(const u16x8*)&Atile[m * 32 + koff];
            u16x8 hi = *(const u16x8*)&Atile[m * 32 + koff + 16];
            afr[i] = combine_bf16(lo, hi);
        }
        // B fragment (same as A): lanes 0-15 -> K {0..7,16..23}, lanes 16-31 -> K {8..15,24..31}.
        bf16x16 bfr[4];
#pragma unroll
        for (int j = 0; j < 4; ++j) {
            int n    = wn + j * 16 + l16;
            int koff = half * 8;
            u16x8 lo = *(const u16x8*)&Btile[n * 32 + koff];
            u16x8 hi = *(const u16x8*)&Btile[n * 32 + koff + 16];
            bfr[j] = combine_bf16(lo, hi);
        }
#pragma unroll
        for (int i = 0; i < 2; ++i)
#pragma unroll
            for (int j = 0; j < 4; ++j)
                acc[i][j] = wmma_bf16(afr[i], bfr[j], acc[i][j]);
        __syncthreads();
    }

    // ---- epilogue: bias add + store (C layout: VGPR r -> M = half*8+r) --
#pragma unroll
    for (int j = 0; j < 4; ++j) {
        int   n  = bn + wn + j * 16 + l16;
        float bv = bias[n];
#pragma unroll
        for (int i = 0; i < 2; ++i) {
#pragma unroll
            for (int r = 0; r < 8; ++r) {
                int   m   = bm + wm + i * 16 + half * 8 + r;
                float val = acc[i][j][r] + bv;
                if (OUT_BF16)
                    ((unsigned short*)Cv)[(size_t)m * N + n] = f32_to_bf16(val);
                else
                    ((float*)Cv)[(size_t)m * N + n] = val;
            }
        }
    }
}

// ---------------------------------------------------------------------------
// Per-token attention over heads: for each token b,
//   S = q_b(16x64) k_b^T / 8 ; P = softmax_rows(S) ; O = P v_b (16x64)
// One wave per token, all matmuls via v_wmma_f32_16x16x32_bf16.
// ---------------------------------------------------------------------------
__global__ __launch_bounds__(256) void attn_kernel(
    const unsigned short* __restrict__ q, const unsigned short* __restrict__ k,
    const unsigned short* __restrict__ v, unsigned short* __restrict__ o) {
    __shared__ unsigned short Psm[8][16 * 32];  // P, K-padded to 32 with zeros
    __shared__ unsigned short Vtm[8][64 * 16];  // v^T : [d][g]

    const int wid  = threadIdx.x >> 5;
    const int lane = threadIdx.x & 31;
    const int half = lane >> 4;
    const int l16  = lane & 15;
    const int tok  = blockIdx.x * 8 + wid;

    const unsigned short* qb = q + (size_t)tok * 1024;
    const unsigned short* kb = k + (size_t)tok * 1024;
    const unsigned short* vb = v + (size_t)tok * 1024;
    unsigned short*       ob = o + (size_t)tok * 1024;

    // ---- S = q k^T  (contraction over d = 64 -> two K=32 WMMAs) ----------
    v8f s = (v8f){0.f,0.f,0.f,0.f,0.f,0.f,0.f,0.f};
#pragma unroll
    for (int f = 0; f < 2; ++f) {
        int koff = half * 8;  // A layout
        u16x8 alo = *(const u16x8*)(qb + l16 * 64 + f * 32 + koff);
        u16x8 ahi = *(const u16x8*)(qb + l16 * 64 + f * 32 + koff + 16);
        bf16x16 af  = combine_bf16(alo, ahi);
        // B = k^T: B[kidx][n] = k_b[n][kidx] -> row n of k is contiguous
        u16x8 blo = *(const u16x8*)(kb + l16 * 64 + f * 32 + half * 8);
        u16x8 bhi = *(const u16x8*)(kb + l16 * 64 + f * 32 + half * 8 + 16);
        bf16x16 bf  = combine_bf16(blo, bhi);
        s = wmma_bf16(af, bf, s);
    }

    // ---- softmax over columns (per row; row spread across 16 lanes) ------
    float p[8];
#pragma unroll
    for (int r = 0; r < 8; ++r) {
        float x  = s[r] * 0.125f;  // 1/sqrt(64)
        float mx = x;
#pragma unroll
        for (int d = 1; d < 16; d <<= 1) mx = fmaxf(mx, __shfl_xor(mx, d, 16));
        float e  = __expf(x - mx);
        float sm = e;
#pragma unroll
        for (int d = 1; d < 16; d <<= 1) sm += __shfl_xor(sm, d, 16);
        p[r] = e / sm;
    }

    // ---- stage P (zero-padded K) and v^T into LDS ------------------------
    unsigned short* Pw = Psm[wid];
#pragma unroll
    for (int r = 0; r < 8; ++r) {
        int m = half * 8 + r;
        Pw[m * 32 + l16]      = f32_to_bf16(p[r]);
        Pw[m * 32 + 16 + l16] = 0;
    }
    unsigned short* Vw = Vtm[wid];
#pragma unroll
    for (int c = 0; c < 4; ++c) {
        int idx = lane * 32 + c * 8;
        u16x8 val = *(const u16x8*)(vb + idx);
        int g = idx >> 6, d = idx & 63;
#pragma unroll
        for (int j = 0; j < 8; ++j) Vw[(d + j) * 16 + g] = val[j];
    }
    __syncthreads();

    // ---- O = P v : 4 column tiles of 16, K = 16 (padded to 32) -----------
    {
        int koff = half * 8;
        u16x8 alo = *(const u16x8*)&Pw[l16 * 32 + koff];
        u16x8 ahi = *(const u16x8*)&Pw[l16 * 32 + koff + 16];  // zeros
        bf16x16 af  = combine_bf16(alo, ahi);
#pragma unroll
        for (int t = 0; t < 4; ++t) {
            u16x16 tv = (u16x16)0;
            {  // elements 0-7 carry K=half*8..+7; elements 8-15 (K=16..31) are zero-padded
                u16x8 lo = *(const u16x8*)&Vw[(t * 16 + l16) * 16 + half * 8];
#pragma unroll
                for (int j = 0; j < 8; ++j) { tv[j] = lo[j]; }
            }
            bf16x16 bf = __builtin_bit_cast(bf16x16, tv);
            v8f oacc = (v8f){0.f,0.f,0.f,0.f,0.f,0.f,0.f,0.f};
            oacc = wmma_bf16(af, bf, oacc);
#pragma unroll
            for (int r = 0; r < 8; ++r) {
                int m = half * 8 + r;
                ob[m * 64 + t * 16 + l16] = f32_to_bf16(oacc[r]);
            }
        }
    }
}

// ---------------------------------------------------------------------------
// Launch
// ---------------------------------------------------------------------------
extern "C" void kernel_launch(void* const* d_in, const int* in_sizes, int n_in,
                              void* d_out, int out_size, void* d_ws, size_t ws_size,
                              hipStream_t stream) {
    (void)in_sizes; (void)n_in; (void)out_size; (void)ws_size;
    constexpr int NB = 32768, DM = 1024;

    const float* queries = (const float*)d_in[0];
    const float* keys    = (const float*)d_in[1];
    const float* values  = (const float*)d_in[2];
    const float* Wq = (const float*)d_in[3];
    const float* bq = (const float*)d_in[4];
    const float* Wk = (const float*)d_in[5];
    const float* bk = (const float*)d_in[6];
    const float* Wv = (const float*)d_in[7];
    const float* bv = (const float*)d_in[8];
    const float* Wo = (const float*)d_in[9];
    const float* bo = (const float*)d_in[10];
    float* out = (float*)d_out;

    // workspace layout (bf16 buffers)
    unsigned short* qs   = (unsigned short*)d_ws;
    unsigned short* ks   = qs   + (size_t)NB * DM;
    unsigned short* vs   = ks   + (size_t)NB * DM;
    unsigned short* at   = vs   + (size_t)NB * DM;
    unsigned short* wq16 = at   + (size_t)NB * DM;   // transposed bf16 weights
    unsigned short* wk16 = wq16 + (size_t)DM * DM;
    unsigned short* wv16 = wk16 + (size_t)DM * DM;
    unsigned short* wo16 = wv16 + (size_t)DM * DM;

    // 1) weights fp32 -> bf16, transposed to [n][k]
    {
        dim3 g(DM / 32, DM / 8), b(256);
        cvt_w_transpose_kernel<<<g, b, 0, stream>>>(Wq, wq16);
        cvt_w_transpose_kernel<<<g, b, 0, stream>>>(Wk, wk16);
        cvt_w_transpose_kernel<<<g, b, 0, stream>>>(Wv, wv16);
        cvt_w_transpose_kernel<<<g, b, 0, stream>>>(Wo, wo16);
    }

    // 2) Q/K/V projections: fp32 activations x bf16 weights -> bf16
    {
        dim3 g(NB / 128, DM / 128), b(256);
        gemm_bf16_kernel<true, true><<<g, b, 0, stream>>>(queries, wq16, bq, qs, NB, DM, DM);
        gemm_bf16_kernel<true, true><<<g, b, 0, stream>>>(keys,    wk16, bk, ks, NB, DM, DM);
        gemm_bf16_kernel<true, true><<<g, b, 0, stream>>>(values,  wv16, bv, vs, NB, DM, DM);
    }

    // 3) per-token attention over heads (WMMA)
    {
        dim3 g(NB / 8), b(256);
        attn_kernel<<<g, b, 0, stream>>>(qs, ks, vs, at);
    }

    // 4) output projection: bf16 x bf16 -> fp32 d_out (A staged via async DMA)
    {
        dim3 g(NB / 128, DM / 128), b(256);
        gemm_bf16_kernel<false, false><<<g, b, 0, stream>>>(at, wo16, bo, out, NB, DM, DM);
    }
}
